// RNN_84396107366550
// MI455X (gfx1250) — matrix operand order repaired and checked by reading, not compiled, not run on hardware
//
#include <hip/hip_runtime.h>
#include <hip/hip_bf16.h>
#include <math.h>

// ---------------------------------------------------------------------------
// Fused 3-layer Elman RNN for MI455X (gfx1250, wave32, WMMA f16 16x16x32).
// One wave owns a 16-row batch tile for all 512 timesteps and all layers.
// Weights pre-swizzled into WMMA B-fragment layout in LDS; hidden state
// cycles accumulators -> v_tanh_f32 -> LDS (f16) -> A-fragments; HBM is
// touched only for x indices (in, software-pipelined) and logits (out).
// ---------------------------------------------------------------------------

typedef __attribute__((ext_vector_type(16))) _Float16 v16h;
typedef __attribute__((ext_vector_type(8)))  _Float16 v8h;
typedef __attribute__((ext_vector_type(8)))  float    v8f;

#define T_SEQ  512
#define BATCHN 2048
#define VOCABN 27
#define EMBN   20
#define NOUTN  26
#define NFRAG  36

// wave-local LDS fence: DS ops are in-order per wave; this is a scheduling
// barrier so the compiler never hoists the A-fragment reloads above the
// state stores (and vice versa).
#define LDS_FENCE() asm volatile("s_wait_dscnt 0x0" ::: "memory")

#define WMMA(a, b, c) \
  __builtin_amdgcn_wmma_f32_16x16x32_f16(false, (a), false, (b), (short)0, (c), false, false)

// B fragment f, this lane's 16 halves (2x ds_load_b128)
#define BF(f) (*(const v16h*)&sB[(f) * 512 + lane * 16])

// Hardware tanh (CDNA5 V_TANH_F32, a TRANS32 op) instead of libm's branchy
// exp-based expansion — the activation count (80/step) otherwise dominates
// the serial recurrence latency.
__device__ __forceinline__ float fast_tanh(float x) {
#if __has_builtin(__builtin_amdgcn_tanhf)
  return __builtin_amdgcn_tanhf(x);
#elif __has_builtin(__builtin_amdgcn_tanh_f32)
  return __builtin_amdgcn_tanh_f32(x);
#else
  float r;
  // v_nop covers the TRANS32 result-use hazard the compiler can't see.
  asm("v_tanh_f32 %0, %1\n\tv_nop" : "=v"(r) : "v"(x));
  return r;
#endif
}

// A fragment (16xK f16, ISA 7.12.2 layout): lane L -> row m = L&15,
// half = L>>4 ; elems 0..7 = K (kc*32 + 8*half + i), elems 8..15 = +16.
__device__ __forceinline__ v16h lds_a(const _Float16* base, int strideH, int kc, int lane) {
  const int m = lane & 15, ah = lane >> 4;
  const _Float16* p = base + m * strideH + kc * 32 + ah * 8;
  v8h lo = *(const v8h*)(p);
  v8h hi = *(const v8h*)(p + 16);
  v16h a;
#pragma unroll
  for (int i = 0; i < 8; ++i) { a[i] = lo[i]; a[i + 8] = hi[i]; }
  return a;
}

// Embedding A fragment: per-lane row = emb row of this lane's batch element.
__device__ __forceinline__ v16h lds_a_emb(const _Float16* embLds, int idx, int lane) {
  const int ah = lane >> 4;
  const _Float16* p = embLds + idx * 32 + ah * 8;
  v8h lo = *(const v8h*)(p);
  v8h hi = *(const v8h*)(p + 16);
  v16h a;
#pragma unroll
  for (int i = 0; i < 8; ++i) { a[i] = lo[i]; a[i + 8] = hi[i]; }
  return a;
}

// accumulator D layout: elem r, lane L -> m = r + 8*(L>>4), n = L&15
__device__ __forceinline__ void store_act(_Float16* base, int strideH, int nt, int lane,
                                          const v8f& acc, float bias) {
  const int n  = nt * 16 + (lane & 15);
  const int mh = (lane >> 4) * 8;
#pragma unroll
  for (int r = 0; r < 8; ++r)
    base[(mh + r) * strideH + n] = (_Float16)fast_tanh(acc[r] + bias);
}

__global__ __launch_bounds__(128) void rnn_fused_gfx1250(
    const int*   __restrict__ x,    const float* __restrict__ emb,
    const float* __restrict__ Wih1, const float* __restrict__ Whh1,
    const float* __restrict__ bih1, const float* __restrict__ bhh1,
    const float* __restrict__ Wih2, const float* __restrict__ Whh2,
    const float* __restrict__ bih2, const float* __restrict__ bhh2,
    const float* __restrict__ Wih3, const float* __restrict__ Whh3,
    const float* __restrict__ bih3, const float* __restrict__ bhh3,
    const float* __restrict__ Wfc,  const float* __restrict__ bfc,
    float* __restrict__ out)
{
  // B fragments: [frag][lane][16 halves]  (36 KB)
  __shared__ _Float16 sB[NFRAG * 512];
  // Embedding table, f16, K padded 20->32 with zeros (1.7 KB)
  __shared__ _Float16 sEmb[VOCABN * 32];
  // Per-wave activation staging: h1 16x32 @0, h2 16x64 @512, h3 16x64 @1536
  __shared__ _Float16 sAct[4][2560];

  const int tid  = threadIdx.x;
  const int lane = tid & 31;
  const int wave = tid >> 5;

  // ---- one-time prep: swizzle weights into B-fragment layout -------------
  // B[k][n] = W[n][k] (zero-padded).  Fragment lane l: n = nb + (l&15),
  // K group = 8*(l>>4); halves i<8 -> k = kb + 8*(l>>4) + i, i>=8 -> +8 more.
  for (int slot = tid; slot < NFRAG * 32; slot += 128) {
    const int f = slot >> 5, l = slot & 31;
    const float* W; int nb, kb, rows, cols;
    if      (f < 2)  { W = Wih1; nb = f * 16;        kb = 0;            rows = 32; cols = 20; }
    else if (f < 4)  { W = Whh1; nb = (f - 2) * 16;  kb = 0;            rows = 32; cols = 32; }
    else if (f < 8)  { W = Wih2; nb = (f - 4) * 16;  kb = 0;            rows = 64; cols = 32; }
    else if (f < 16) { int g = f - 8;  W = Whh2; nb = (g >> 1) * 16; kb = (g & 1) * 32; rows = 64; cols = 64; }
    else if (f < 24) { int g = f - 16; W = Wih3; nb = (g >> 1) * 16; kb = (g & 1) * 32; rows = 64; cols = 64; }
    else if (f < 32) { int g = f - 24; W = Whh3; nb = (g >> 1) * 16; kb = (g & 1) * 32; rows = 64; cols = 64; }
    else             { int g = f - 32; W = Wfc;  nb = (g >> 1) * 16; kb = (g & 1) * 32; rows = 26; cols = 64; }
    const int n  = nb + (l & 15);
    const int kh = (l >> 4) * 8;
    _Float16* dst = &sB[slot * 16];
    for (int i = 0; i < 16; ++i) {
      const int k = kb + kh + (i < 8 ? i : i + 8);
      const float v = (k < cols && n < rows) ? W[n * cols + k] : 0.0f;
      dst[i] = (_Float16)v;
    }
  }
  for (int s = tid; s < VOCABN * 32; s += 128) {
    const int r = s >> 5, c = s & 31;
    sEmb[s] = (c < EMBN) ? (_Float16)emb[r * EMBN + c] : (_Float16)0.0f;
  }
  for (int s = lane; s < 2560; s += 32) sAct[wave][s] = (_Float16)0.0f;  // h0 = 0
  __syncthreads();

  _Float16* h1w = &sAct[wave][0];
  _Float16* h2w = &sAct[wave][512];
  _Float16* h3w = &sAct[wave][1536];

  const int bb = blockIdx.x * 64 + wave * 16;   // batch tile base (16 rows)
  const int n  = lane & 15;

  // combined biases, held in registers (indexed by this lane's N column)
  float b1r[2], b2r[4], b3r[4], bfr[2];
#pragma unroll
  for (int q = 0; q < 2; ++q) b1r[q] = bih1[q * 16 + n] + bhh1[q * 16 + n];
#pragma unroll
  for (int q = 0; q < 4; ++q) {
    b2r[q] = bih2[q * 16 + n] + bhh2[q * 16 + n];
    b3r[q] = bih3[q * 16 + n] + bhh3[q * 16 + n];
  }
  bfr[0] = bfc[n];
  bfr[1] = (16 + n < NOUTN) ? bfc[16 + n] : 0.0f;

  // software-pipelined x index: the load for t+1 is issued at the top of
  // iteration t and only waited on next iteration, hiding the global-load
  // latency behind a full timestep of WMMA work on the serial chain.
  int idx_cur = x[bb + n];
  for (int t = 0; t < T_SEQ; ++t) {
    const int tn = (t + 1 < T_SEQ) ? (t + 1) : t;        // clamp, no branch
    const int idx_nxt = x[tn * BATCHN + bb + n];

    // ---- layer 1: h1 = tanh(e @ Wih1^T + h1 @ Whh1^T + b) ----------------
    v16h aE  = lds_a_emb(sEmb, idx_cur, lane);      // 16x32 (K padded)
    v16h aH1 = lds_a(h1w, 32, 0, lane);             // previous h1, 16x32
    v8f c0 = {}, c1 = {};
    c0 = WMMA(aE,  BF(0), c0);  c0 = WMMA(aH1, BF(2), c0);
    c1 = WMMA(aE,  BF(1), c1);  c1 = WMMA(aH1, BF(3), c1);
    LDS_FENCE();
    store_act(h1w, 32, 0, lane, c0, b1r[0]);
    store_act(h1w, 32, 1, lane, c1, b1r[1]);
    LDS_FENCE();

    // ---- layer 2: K_ih = 32 (h1), K_hh = 64 ------------------------------
    v16h aH1n = lds_a(h1w, 32, 0, lane);
    v16h aH2a = lds_a(h2w, 64, 0, lane);
    v16h aH2b = lds_a(h2w, 64, 1, lane);
    v8f d0 = {}, d1 = {}, d2 = {}, d3 = {};
    d0 = WMMA(aH1n, BF(4), d0); d0 = WMMA(aH2a, BF(8),  d0); d0 = WMMA(aH2b, BF(9),  d0);
    d1 = WMMA(aH1n, BF(5), d1); d1 = WMMA(aH2a, BF(10), d1); d1 = WMMA(aH2b, BF(11), d1);
    d2 = WMMA(aH1n, BF(6), d2); d2 = WMMA(aH2a, BF(12), d2); d2 = WMMA(aH2b, BF(13), d2);
    d3 = WMMA(aH1n, BF(7), d3); d3 = WMMA(aH2a, BF(14), d3); d3 = WMMA(aH2b, BF(15), d3);
    LDS_FENCE();
    store_act(h2w, 64, 0, lane, d0, b2r[0]);
    store_act(h2w, 64, 1, lane, d1, b2r[1]);
    store_act(h2w, 64, 2, lane, d2, b2r[2]);
    store_act(h2w, 64, 3, lane, d3, b2r[3]);
    LDS_FENCE();

    // ---- layer 3: K_ih = 64, K_hh = 64 -----------------------------------
    v16h aH2na = lds_a(h2w, 64, 0, lane);
    v16h aH2nb = lds_a(h2w, 64, 1, lane);
    v16h aH3a  = lds_a(h3w, 64, 0, lane);
    v16h aH3b  = lds_a(h3w, 64, 1, lane);
    v8f e0 = {}, e1 = {}, e2 = {}, e3 = {};
    e0 = WMMA(aH2na, BF(16), e0); e0 = WMMA(aH2nb, BF(17), e0);
    e0 = WMMA(aH3a,  BF(24), e0); e0 = WMMA(aH3b,  BF(25), e0);
    e1 = WMMA(aH2na, BF(18), e1); e1 = WMMA(aH2nb, BF(19), e1);
    e1 = WMMA(aH3a,  BF(26), e1); e1 = WMMA(aH3b,  BF(27), e1);
    e2 = WMMA(aH2na, BF(20), e2); e2 = WMMA(aH2nb, BF(21), e2);
    e2 = WMMA(aH3a,  BF(28), e2); e2 = WMMA(aH3b,  BF(29), e2);
    e3 = WMMA(aH2na, BF(22), e3); e3 = WMMA(aH2nb, BF(23), e3);
    e3 = WMMA(aH3a,  BF(30), e3); e3 = WMMA(aH3b,  BF(31), e3);
    LDS_FENCE();
    store_act(h3w, 64, 0, lane, e0, b3r[0]);
    store_act(h3w, 64, 1, lane, e1, b3r[1]);
    store_act(h3w, 64, 2, lane, e2, b3r[2]);
    store_act(h3w, 64, 3, lane, e3, b3r[3]);
    LDS_FENCE();

    // ---- FC head: out = h3 @ Wfc^T + b_fc  (N = 26, padded to 32) --------
    v16h aH3na = lds_a(h3w, 64, 0, lane);
    v16h aH3nb = lds_a(h3w, 64, 1, lane);
    v8f f0 = {}, f1 = {};
    f0 = WMMA(aH3na, BF(32), f0); f0 = WMMA(aH3nb, BF(33), f0);
    f1 = WMMA(aH3na, BF(34), f1); f1 = WMMA(aH3nb, BF(35), f1);
    {
      const int  mh = (lane >> 4) * 8;
      const long rowbase = (long)t * BATCHN + bb;
#pragma unroll
      for (int r = 0; r < 8; ++r) {
        const long row = rowbase + mh + r;
        out[row * NOUTN + n] = f0[r] + bfr[0];
        if (16 + n < NOUTN) out[row * NOUTN + 16 + n] = f1[r] + bfr[1];
      }
    }
    idx_cur = idx_nxt;
  }
}

extern "C" void kernel_launch(void* const* d_in, const int* in_sizes, int n_in,
                              void* d_out, int out_size, void* d_ws, size_t ws_size,
                              hipStream_t stream) {
  (void)in_sizes; (void)n_in; (void)out_size; (void)d_ws; (void)ws_size;
  const int*   x    = (const int*)d_in[0];
  const float* emb  = (const float*)d_in[1];
  const float* Wih1 = (const float*)d_in[2];
  const float* Whh1 = (const float*)d_in[3];
  const float* bih1 = (const float*)d_in[4];
  const float* bhh1 = (const float*)d_in[5];
  const float* Wih2 = (const float*)d_in[6];
  const float* Whh2 = (const float*)d_in[7];
  const float* bih2 = (const float*)d_in[8];
  const float* bhh2 = (const float*)d_in[9];
  const float* Wih3 = (const float*)d_in[10];
  const float* Whh3 = (const float*)d_in[11];
  const float* bih3 = (const float*)d_in[12];
  const float* bhh3 = (const float*)d_in[13];
  const float* Wfc  = (const float*)d_in[14];
  const float* bfc  = (const float*)d_in[15];
  float* out = (float*)d_out;

  // 2048 batch rows / (4 waves * 16 rows) = 32 workgroups of 128 threads
  rnn_fused_gfx1250<<<dim3(32), dim3(128), 0, stream>>>(
      x, emb, Wih1, Whh1, bih1, bhh1, Wih2, Whh2, bih2, bhh2,
      Wih3, Whh3, bih3, bhh3, Wfc, bfc, out);
}
